// Speller_69269232550072
// MI455X (gfx1250) — matrix operand order repaired and checked by reading, not compiled, not run on hardware
//
#include <hip/hip_runtime.h>
#include <hip/hip_bf16.h>

// ---------------------------------------------------------------------------
// Speller (attention-LSTM decoder) for MI455X / gfx1250.
// Strategy: one-time fp32->bf16 weight conversion (halves L2 traffic; whole
// model fits in 192MB L2), then 129 sequential decoder steps, each a short
// pipeline of WMMA bf16 GEMM kernels (M=32 tiles -> 32x16 per wave, two
// v_wmma_f32_16x16x32_bf16 sharing one B operand) plus small VALU kernels
// for softmax-attention and the LSTM cell nonlinearity.
// ---------------------------------------------------------------------------

typedef __attribute__((ext_vector_type(16))) __bf16 v16bf;
typedef __attribute__((ext_vector_type(8)))  __bf16 v8bf;
typedef __attribute__((ext_vector_type(8)))  float  v8f;

#define B_    32
#define T_    500
#define DENC_ 512
#define H_    1024
#define EMB_  512
#define ODIM_ 5000
#define ATT_  512
#define LY1_  129          // LY + 1 decoder steps
#define EOS_  4999
#define NPADPROJ_ 5008     // ODIM padded to multiple of 16

// ---------------------------------------------------------------------------
// WMMA operand loaders (layouts per CDNA5 ISA §7.12.2, wave32)
// ---------------------------------------------------------------------------

// A-matrix 16x32 bf16 chunk for one lane.
// lanes 0-15 (row M=lane):   elems 0..7 = K kc+0..7,  elems 8..15 = K kc+16..23
// lanes 16-31 (row M=lane-16): elems 0..7 = K kc+8..15, elems 8..15 = K kc+24..31
__device__ __forceinline__ v16bf lda_chunk(const __bf16* __restrict__ row,
                                           int kc, int koffA) {
  v8bf lo = *(const v8bf*)(row + kc + koffA);
  v8bf hi = *(const v8bf*)(row + kc + koffA + 16);
  return __builtin_shufflevector(lo, hi, 0,1,2,3,4,5,6,7,8,9,10,11,12,13,14,15);
}

// B-matrix 32x16 bf16 chunk: lane holds column N=lane&15; lanes 0-15 K kc+0..7,kc+16..23,
// lanes 16-31 K kc+8..15,kc+24..31 (weights stored (N,K) row-major -> contiguous K).
__device__ __forceinline__ v16bf ldb_chunk(const __bf16* __restrict__ wrow,
                                           int kc, int koffB) {
  v8bf lo = *(const v8bf*)(wrow + kc + koffB);
  v8bf hi = *(const v8bf*)(wrow + kc + koffB + 16);
  return __builtin_shufflevector(lo, hi, 0,1,2,3,4,5,6,7,8,9,10,11,12,13,14,15);
}

// ---------------------------------------------------------------------------
// Generic bf16 WMMA GEMM:  C[M x Nreal] = A1[M x K1] @ W1[N x K1]^T
//                                       (+ A2[M x K2] @ W2[N x K2]^T)
//                                       (+ biasA[n] + biasB[n])
// M is a multiple of 32; one wave computes a 32x16 tile (2 WMMAs / K-chunk,
// B operand shared). Weights are (N,K) row-major bf16.
// ---------------------------------------------------------------------------
__global__ void wmma_gemm_kernel(const __bf16* __restrict__ A1,
                                 const __bf16* __restrict__ W1, int K1,
                                 const __bf16* __restrict__ A2,
                                 const __bf16* __restrict__ W2, int K2,
                                 const float* __restrict__ biasA,
                                 const float* __restrict__ biasB,
                                 float* __restrict__ C, long ldc,
                                 int Nreal, int Ntiles, int Mpairs)
{
  int wave = (int)((blockIdx.x * blockDim.x + threadIdx.x) >> 5);
  if (wave >= Ntiles * Mpairs) return;          // whole-wave uniform exit
  int lane  = threadIdx.x & 31;
  int ntile = wave % Ntiles;
  int mpair = wave / Ntiles;
  int n0 = ntile * 16;
  int m0 = mpair * 32;
  int nlane = n0 + (lane & 15);
  int koffA = (lane < 16) ? 0 : 8;
  int koffB = (lane < 16) ? 0 : 8;

  v8f acc0 = {};
  v8f acc1 = {};

  {
    const __bf16* a0r = A1 + (long)(m0 + (lane & 15)) * K1;
    const __bf16* a1r = a0r + 16L * K1;
    const __bf16* wr  = W1 + (long)nlane * K1;
    for (int kc = 0; kc < K1; kc += 32) {
      v16bf bm = ldb_chunk(wr, kc, koffB);
      v16bf a0 = lda_chunk(a0r, kc, koffA);
      v16bf a1 = lda_chunk(a1r, kc, koffA);
      acc0 = __builtin_amdgcn_wmma_f32_16x16x32_bf16(false, a0, false, bm,
                                                     (short)0, acc0, false, false);
      acc1 = __builtin_amdgcn_wmma_f32_16x16x32_bf16(false, a1, false, bm,
                                                     (short)0, acc1, false, false);
    }
  }
  if (A2 != nullptr) {                          // kernel-arg: wave uniform
    const __bf16* a0r = A2 + (long)(m0 + (lane & 15)) * K2;
    const __bf16* a1r = a0r + 16L * K2;
    const __bf16* wr  = W2 + (long)nlane * K2;
    for (int kc = 0; kc < K2; kc += 32) {
      v16bf bm = ldb_chunk(wr, kc, koffB);
      v16bf a0 = lda_chunk(a0r, kc, koffA);
      v16bf a1 = lda_chunk(a1r, kc, koffA);
      acc0 = __builtin_amdgcn_wmma_f32_16x16x32_bf16(false, a0, false, bm,
                                                     (short)0, acc0, false, false);
      acc1 = __builtin_amdgcn_wmma_f32_16x16x32_bf16(false, a1, false, bm,
                                                     (short)0, acc1, false, false);
    }
  }

  bool nok = nlane < Nreal;
  float bias = 0.f;
  if (nok) {
    if (biasA) bias += biasA[nlane];
    if (biasB) bias += biasB[nlane];
  }
  int mo = (lane < 16) ? 0 : 8;                 // C layout: VGPR r -> row m0+mo+r
  if (nok) {
#pragma unroll
    for (int r = 0; r < 8; ++r) {
      C[(long)(m0 + mo + r)      * ldc + nlane] = acc0[r] + bias;
      C[(long)(m0 + 16 + mo + r) * ldc + nlane] = acc1[r] + bias;
    }
  }
}

// ---------------------------------------------------------------------------
// Attention: e=v.tanh(pre+dec), mask, softmax over T, ctx=w@hs; builds
// x = [ctx ; emb_step] in bf16. One block per batch element.
// ---------------------------------------------------------------------------
__global__ void attn_ctx_kernel(const float* __restrict__ pre,
                                const float* __restrict__ dec,
                                const float* __restrict__ vatt,
                                const int*   __restrict__ hlens,
                                const float* __restrict__ hs,
                                const __bf16* __restrict__ embs, int step,
                                __bf16* __restrict__ xbf)
{
  __shared__ float sdec[ATT_];
  __shared__ float sv[ATT_];
  __shared__ float se[T_ + 12];
  __shared__ float red[256];

  int b = blockIdx.x;
  int tid = threadIdx.x;
  for (int i = tid; i < ATT_; i += 256) { sdec[i] = dec[b * ATT_ + i]; sv[i] = vatt[i]; }
  __syncthreads();

  int hl = hlens[b];
  float myMax = -3.0e38f;
  for (int t = tid; t < T_; t += 256) {
    const float* p = pre + ((long)b * T_ + t) * ATT_;
    float s = 0.f;
    for (int a = 0; a < ATT_; ++a) s += sv[a] * tanhf(p[a] + sdec[a]);
    if (t >= hl) s = -1e9f;
    se[t] = s;
    myMax = fmaxf(myMax, s);
  }
  red[tid] = myMax; __syncthreads();
  for (int o = 128; o > 0; o >>= 1) { if (tid < o) red[tid] = fmaxf(red[tid], red[tid + o]); __syncthreads(); }
  float mx = red[0]; __syncthreads();

  float mySum = 0.f;
  for (int t = tid; t < T_; t += 256) { float w = __expf(se[t] - mx); se[t] = w; mySum += w; }
  red[tid] = mySum; __syncthreads();
  for (int o = 128; o > 0; o >>= 1) { if (tid < o) red[tid] += red[tid + o]; __syncthreads(); }
  float inv = 1.f / red[0]; __syncthreads();

  for (int d = tid; d < DENC_; d += 256) {
    const float* hp = hs + (long)b * T_ * DENC_ + d;
    float acc = 0.f;
    for (int t = 0; t < T_; ++t) acc += se[t] * hp[(long)t * DENC_];
    xbf[b * (DENC_ + EMB_) + d] = (__bf16)(acc * inv);
  }
  for (int d = tid; d < EMB_; d += 256)
    xbf[b * (DENC_ + EMB_) + DENC_ + d] = embs[((long)b * LY1_ + step) * EMB_ + d];
}

// ---------------------------------------------------------------------------
// LSTM cell nonlinearity (biases already folded into g by the GEMM).
// ---------------------------------------------------------------------------
__device__ __forceinline__ float sigm(float x) { return 1.f / (1.f + __expf(-x)); }

__global__ void lstm_cell_kernel(const float* __restrict__ g,
                                 float* __restrict__ c,
                                 __bf16* __restrict__ hbf)
{
  int idx = blockIdx.x * blockDim.x + threadIdx.x;
  if (idx >= B_ * H_) return;
  int b = idx >> 10, j = idx & (H_ - 1);
  const float* gb = g + (long)b * 4 * H_;
  float gi = gb[j], gf = gb[H_ + j], gg = gb[2 * H_ + j], go = gb[3 * H_ + j];
  float cn = sigm(gf) * c[idx] + sigm(gi) * tanhf(gg);
  float hn = sigm(go) * tanhf(cn);
  c[idx]   = cn;
  hbf[idx] = (__bf16)hn;
}

// ---------------------------------------------------------------------------
// Setup kernels
// ---------------------------------------------------------------------------
__global__ void zero_state_kernel(__bf16* hbf, float* c) {
  int i = blockIdx.x * blockDim.x + threadIdx.x;
  if (i < 2 * B_ * H_) { hbf[i] = (__bf16)0.f; c[i] = 0.f; }
}

__global__ void cvt_bf16_kernel(const float* __restrict__ s, __bf16* __restrict__ d, long n) {
  long i = (long)blockIdx.x * blockDim.x + threadIdx.x;
  if (i < n) d[i] = (__bf16)s[i];
}

// src is (K,N) fp32; dst is (N,K) bf16 -> contiguous-K weight rows for WMMA B.
__global__ void cvt_transpose_bf16_kernel(const float* __restrict__ s,
                                          __bf16* __restrict__ d, int K, int N) {
  long i = (long)blockIdx.x * blockDim.x + threadIdx.x;
  if (i < (long)K * N) {
    int k = (int)(i / N), n = (int)(i % N);
    d[(long)n * K + k] = (__bf16)s[i];
  }
}

__global__ void cvt_pad_bf16_kernel(const float* __restrict__ s, __bf16* __restrict__ d,
                                    int rows_real, int rows_pad, int cols) {
  long i = (long)blockIdx.x * blockDim.x + threadIdx.x;
  if (i < (long)rows_pad * cols) {
    int r = (int)(i / cols), cidx = (int)(i % cols);
    d[i] = (r < rows_real) ? (__bf16)s[(long)r * cols + cidx] : (__bf16)0.f;
  }
}

__global__ void tokens_kernel(const int* __restrict__ ys_out,
                              int* __restrict__ ys_in, int* __restrict__ ysref) {
  int idx = blockIdx.x * blockDim.x + threadIdx.x;
  if (idx >= B_ * LY1_) return;
  int b = idx / LY1_, s = idx % LY1_;
  int tok;
  if (s == 0) tok = EOS_;                       // SOS == EOS id
  else { tok = ys_out[b * (LY1_ - 1) + s - 1]; if (tok < 0) tok = EOS_; }
  ys_in[idx] = tok;
  ysref[idx] = (s < LY1_ - 1) ? ys_out[b * (LY1_ - 1) + s] : EOS_;
}

__global__ void embed_kernel(const float* __restrict__ embed_W,
                             const int* __restrict__ ys_in,
                             __bf16* __restrict__ embs) {
  long i = (long)blockIdx.x * blockDim.x + threadIdx.x;
  if (i >= (long)B_ * LY1_ * EMB_) return;
  int d = (int)(i % EMB_);
  long bs = i / EMB_;
  embs[i] = (__bf16)embed_W[(long)ys_in[bs] * EMB_ + d];
}

// ---------------------------------------------------------------------------
// Host orchestration
// ---------------------------------------------------------------------------
extern "C" void kernel_launch(void* const* d_in, const int* in_sizes, int n_in,
                              void* d_out, int out_size, void* d_ws, size_t ws_size,
                              hipStream_t stream)
{
  const float* hs      = (const float*)d_in[0];
  const int*   hlens   = (const int*)  d_in[1];
  const int*   ys_out  = (const int*)  d_in[2];
  const float* embed_W = (const float*)d_in[3];
  const float* Wih     = (const float*)d_in[4];
  const float* Whh     = (const float*)d_in[5];
  const float* bih     = (const float*)d_in[6];
  const float* bhh     = (const float*)d_in[7];
  const float* Wenc    = (const float*)d_in[8];
  const float* Wdec    = (const float*)d_in[9];
  const float* vatt    = (const float*)d_in[10];
  const float* Wproj   = (const float*)d_in[11];
  const float* bproj   = (const float*)d_in[12];

  float* logits = (float*)d_out;                          // [B, LY1, ODIM]
  int*   ysref  = (int*)((float*)d_out + (size_t)B_ * LY1_ * ODIM_);

  // -------- workspace carve-out (256B aligned slices) --------
  char* base = (char*)d_ws;
  size_t off = 0;
  auto carve = [&](size_t bytes) -> char* {
    char* r = base + off;
    off = (off + bytes + 255) & ~(size_t)255;
    return r;
  };
  __bf16* wih_bf   = (__bf16*)carve((size_t)2 * 4 * H_ * (DENC_ + EMB_) * 2);
  __bf16* whh_bf   = (__bf16*)carve((size_t)2 * 4 * H_ * H_ * 2);
  __bf16* wproj_bf = (__bf16*)carve((size_t)NPADPROJ_ * H_ * 2);
  __bf16* wdecT_bf = (__bf16*)carve((size_t)ATT_ * H_ * 2);     // (N=ATT, K=H)
  __bf16* wencT_bf = (__bf16*)carve((size_t)ATT_ * DENC_ * 2);  // (N=ATT, K=DENC)
  __bf16* hs_bf    = (__bf16*)carve((size_t)B_ * T_ * DENC_ * 2);
  float*  pre      = (float*) carve((size_t)B_ * T_ * ATT_ * 4);
  __bf16* embs_bf  = (__bf16*)carve((size_t)B_ * LY1_ * EMB_ * 2);
  __bf16* xbf      = (__bf16*)carve((size_t)B_ * (DENC_ + EMB_) * 2);
  __bf16* hbf      = (__bf16*)carve((size_t)2 * B_ * H_ * 2);   // layers 0,1
  float*  cst      = (float*) carve((size_t)2 * B_ * H_ * 4);
  float*  decb     = (float*) carve((size_t)B_ * ATT_ * 4);
  float*  gbuf     = (float*) carve((size_t)B_ * 4 * H_ * 4);
  int*    ys_in    = (int*)   carve((size_t)B_ * LY1_ * 4);
  (void)ws_size; (void)in_sizes; (void)n_in; (void)out_size;

  auto blks = [](long n, int bs) { return (int)((n + bs - 1) / bs); };

  // -------- one-time setup (re-run every call: deterministic) --------
  zero_state_kernel<<<blks(2 * B_ * H_, 256), 256, 0, stream>>>(hbf, cst);

  long nWih = (long)2 * 4 * H_ * (DENC_ + EMB_);
  cvt_bf16_kernel<<<blks(nWih, 256), 256, 0, stream>>>(Wih, wih_bf, nWih);
  long nWhh = (long)2 * 4 * H_ * H_;
  cvt_bf16_kernel<<<blks(nWhh, 256), 256, 0, stream>>>(Whh, whh_bf, nWhh);
  long nHs = (long)B_ * T_ * DENC_;
  cvt_bf16_kernel<<<blks(nHs, 256), 256, 0, stream>>>(hs, hs_bf, nHs);
  cvt_transpose_bf16_kernel<<<blks((long)DENC_ * ATT_, 256), 256, 0, stream>>>(Wenc, wencT_bf, DENC_, ATT_);
  cvt_transpose_bf16_kernel<<<blks((long)H_ * ATT_, 256), 256, 0, stream>>>(Wdec, wdecT_bf, H_, ATT_);
  cvt_pad_bf16_kernel<<<blks((long)NPADPROJ_ * H_, 256), 256, 0, stream>>>(Wproj, wproj_bf, ODIM_, NPADPROJ_, H_);

  tokens_kernel<<<blks(B_ * LY1_, 256), 256, 0, stream>>>(ys_out, ys_in, ysref);
  embed_kernel<<<blks((long)B_ * LY1_ * EMB_, 256), 256, 0, stream>>>(embed_W, ys_in, embs_bf);

  // pre = hs @ Wenc : M = B*T = 16000 (500 m-pairs), N = 512 (32 tiles), K = 512
  {
    int mpairs = (B_ * T_) / 32, ntiles = ATT_ / 16;
    int waves = mpairs * ntiles;
    wmma_gemm_kernel<<<blks((long)waves * 32, 256), 256, 0, stream>>>(
        hs_bf, wencT_bf, DENC_, nullptr, nullptr, 0, nullptr, nullptr,
        pre, ATT_, ATT_, ntiles, mpairs);
  }

  // -------- sequential decoder steps --------
  for (int step = 0; step < LY1_; ++step) {
    // dec = h0 @ Wdec : M=32, N=512, K=1024
    wmma_gemm_kernel<<<blks(32L * 32, 256), 256, 0, stream>>>(
        hbf, wdecT_bf, H_, nullptr, nullptr, 0, nullptr, nullptr,
        decb, ATT_, ATT_, ATT_ / 16, 1);

    attn_ctx_kernel<<<B_, 256, 0, stream>>>(pre, decb, vatt, hlens, hs,
                                            embs_bf, step, xbf);

    // layer 0 gates: x@Wih0^T + h0@Whh0^T + bih0 + bhh0 : M=32, N=4096, K=1024+1024
    wmma_gemm_kernel<<<blks(256L * 32, 256), 256, 0, stream>>>(
        xbf, wih_bf, DENC_ + EMB_,
        hbf, whh_bf, H_,
        bih, bhh, gbuf, 4 * H_, 4 * H_, (4 * H_) / 16, 1);
    lstm_cell_kernel<<<blks(B_ * H_, 256), 256, 0, stream>>>(gbuf, cst, hbf);

    // layer 1 gates: h0_new@Wih1^T + h1@Whh1^T + biases
    wmma_gemm_kernel<<<blks(256L * 32, 256), 256, 0, stream>>>(
        hbf,            wih_bf + (size_t)4 * H_ * (DENC_ + EMB_), H_,
        hbf + B_ * H_,  whh_bf + (size_t)4 * H_ * H_,             H_,
        bih + 4 * H_, bhh + 4 * H_, gbuf, 4 * H_, 4 * H_, (4 * H_) / 16, 1);
    lstm_cell_kernel<<<blks(B_ * H_, 256), 256, 0, stream>>>(gbuf, cst + B_ * H_,
                                                             hbf + B_ * H_);

    // logits[:, step, :] = h1 @ Wproj^T + bproj : M=32, N=5000 (pad 5008), K=1024
    wmma_gemm_kernel<<<blks((NPADPROJ_ / 16) * 32L, 256), 256, 0, stream>>>(
        hbf + B_ * H_, wproj_bf, H_, nullptr, nullptr, 0,
        bproj, nullptr,
        logits + (size_t)step * ODIM_, (long)LY1_ * ODIM_,
        ODIM_, NPADPROJ_ / 16, 1);
  }
}
